// MultiHeadAttention_81363860455741
// MI455X (gfx1250) — matrix operand order repaired and checked by reading, not compiled, not run on hardware
//
#include <hip/hip_runtime.h>
#include <hip/hip_bf16.h>

typedef __attribute__((ext_vector_type(16))) __bf16 v16bf;
typedef __attribute__((ext_vector_type(8)))  __bf16 v8bf;
typedef __attribute__((ext_vector_type(8)))  float  v8f;
typedef __attribute__((ext_vector_type(4)))  float  v4f;

#define LDSROW 40  // GEMM LDS row stride in halves: 32 data + 8 pad
#define KSTR   72  // attn K-tile LDS row stride (64 data + 8 pad)
#define VSTR   40  // attn V-tile LDS row stride (32 data + 8 pad)
#define PSTR   40  // attn P-slab LDS row stride (32 data + 8 pad)

__device__ __forceinline__ v8bf cvt8(v4f a, v4f b) {
    v8bf r;
    r[0] = (__bf16)a[0]; r[1] = (__bf16)a[1]; r[2] = (__bf16)a[2]; r[3] = (__bf16)a[3];
    r[4] = (__bf16)b[0]; r[5] = (__bf16)b[1]; r[6] = (__bf16)b[2]; r[7] = (__bf16)b[3];
    return r;
}
__device__ __forceinline__ v16bf cat16(v8bf lo, v8bf hi) {
    return __builtin_shufflevector(lo, hi, 0,1,2,3,4,5,6,7,8,9,10,11,12,13,14,15);
}
__device__ __forceinline__ v8f wmma_bf16(v16bf a, v16bf b, v8f c) {
    return __builtin_amdgcn_wmma_f32_16x16x32_bf16(false, a, false, b, (short)0, c, false, false);
}

// ---- VALU (DPP) row reductions across the 16 lanes of a half-wave ----
// C-layout: lanes 0-15 hold rows 0..7, lanes 16-31 hold rows 8..15.
// xor1 = quad_perm[1,0,3,2]; xor2 = quad_perm[2,3,0,1];
// after bits0-1 reduced, row_half_mirror (^7) == xor4; after bits0-2, row_mirror (^15) == xor8.
template<int CTRL>
__device__ __forceinline__ float dpp_mov(float x) {
    return __int_as_float(
        __builtin_amdgcn_mov_dpp(__float_as_int(x), CTRL, 0xF, 0xF, true));
}
__device__ __forceinline__ float hmax16(float x) {
    x = fmaxf(x, dpp_mov<0x0B1>(x));  // quad_perm [1,0,3,2]
    x = fmaxf(x, dpp_mov<0x04E>(x));  // quad_perm [2,3,0,1]
    x = fmaxf(x, dpp_mov<0x141>(x));  // row_half_mirror
    x = fmaxf(x, dpp_mov<0x140>(x));  // row_mirror
    return x;
}
__device__ __forceinline__ float hsum16(float x) {
    x += dpp_mov<0x0B1>(x);
    x += dpp_mov<0x04E>(x);
    x += dpp_mov<0x141>(x);
    x += dpp_mov<0x140>(x);
    return x;
}

// ---- CDNA5 async global->LDS copy (ASYNCcnt-tracked), inline asm ----
#define LDSOFF(p) ((unsigned)(size_t)(p))
__device__ __forceinline__ void async_copy16(unsigned lds_off, const void* gaddr) {
    asm volatile("global_load_async_to_lds_b128 %0, %1, off"
                 :: "v"(lds_off), "v"(gaddr) : "memory");
}
__device__ __forceinline__ void wait_async_le4() {
    asm volatile("s_wait_asynccnt 0x4" ::: "memory");
}

// ============================================================================
// C = A @ W^T ; A: [M,1024] (fp32 or bf16), W: [1024,1024] fp32.
// Block = 256 thr (8 waves), 128x128 tile; wave tile 32x64 (2x4 WMMA tiles).
// STORE_T: output written transposed as bf16 [B][1024][4096].
// ============================================================================
template<int A_BF16, int STORE_T, int OUT_F32>
__global__ __launch_bounds__(256, 2)
void gemm_wmma_kernel(const void* __restrict__ A_, const float* __restrict__ W,
                      void* __restrict__ C_, int M) {
    (void)M;
    constexpr int K = 1024;
    constexpr int N = 1024;
    __shared__ __bf16 As[128 * LDSROW];
    __shared__ __bf16 Bs[128 * LDSROW];

    const int tid  = threadIdx.x;
    const int wave = tid >> 5;
    const int lane = tid & 31;
    const int l16  = lane & 15;
    const int g    = lane >> 4;
    const int bm   = blockIdx.y * 128;
    const int bn   = blockIdx.x * 128;
    const int wm   = (wave >> 1) * 32;
    const int wn   = (wave & 1) * 64;
    const int srow  = tid >> 1;
    const int shalf = (tid & 1) * 16;

    v8f acc[2][4];
#pragma unroll
    for (int i = 0; i < 2; ++i)
#pragma unroll
        for (int j = 0; j < 4; ++j) { v8f z = {}; acc[i][j] = z; }

    const float*  Af = (const float*)A_;
    const __bf16* Ab = (const __bf16*)A_;

    for (int k0 = 0; k0 < K; k0 += 32) {
        if (A_BF16) {
            const v8bf* s = (const v8bf*)(Ab + (size_t)(bm + srow) * K + k0 + shalf);
            v8bf x0 = s[0], x1 = s[1];
            *(v8bf*)&As[srow * LDSROW + shalf]     = x0;
            *(v8bf*)&As[srow * LDSROW + shalf + 8] = x1;
        } else {
            const v4f* s = (const v4f*)(Af + (size_t)(bm + srow) * K + k0 + shalf);
            v4f f0 = s[0], f1 = s[1], f2 = s[2], f3 = s[3];
            *(v8bf*)&As[srow * LDSROW + shalf]     = cvt8(f0, f1);
            *(v8bf*)&As[srow * LDSROW + shalf + 8] = cvt8(f2, f3);
        }
        {
            const v4f* s = (const v4f*)(W + (size_t)(bn + srow) * K + k0 + shalf);
            v4f f0 = s[0], f1 = s[1], f2 = s[2], f3 = s[3];
            *(v8bf*)&Bs[srow * LDSROW + shalf]     = cvt8(f0, f1);
            *(v8bf*)&Bs[srow * LDSROW + shalf + 8] = cvt8(f2, f3);
        }
        if (k0 + 32 < K) {
            __builtin_prefetch(Af + (size_t)(bm + srow) * K + k0 + 32 + shalf, 0, 1);
            __builtin_prefetch(W  + (size_t)(bn + srow) * K + k0 + 32 + shalf, 0, 1);
        }
        __syncthreads();

        v16bf a[2], b[4];
#pragma unroll
        for (int ms = 0; ms < 2; ++ms) {
            const __bf16* p = &As[(wm + ms * 16 + l16) * LDSROW];
            a[ms] = cat16(*(const v8bf*)(p + g * 16), *(const v8bf*)(p + g * 16 + 8));
        }
#pragma unroll
        for (int ns = 0; ns < 4; ++ns) {
            const __bf16* p = &Bs[(wn + ns * 16 + l16) * LDSROW] + g * 16;
            b[ns] = cat16(*(const v8bf*)p, *(const v8bf*)(p + 8));
        }
#pragma unroll
        for (int ms = 0; ms < 2; ++ms)
#pragma unroll
            for (int ns = 0; ns < 4; ++ns)
                acc[ms][ns] = wmma_bf16(a[ms], b[ns], acc[ms][ns]);
        __syncthreads();
    }

#pragma unroll
    for (int ms = 0; ms < 2; ++ms) {
#pragma unroll
        for (int ns = 0; ns < 4; ++ns) {
            const int mb = bm + wm + ms * 16 + g * 8;
            const int n  = bn + wn + ns * 16 + l16;
            if (STORE_T) {
                __bf16* out = (__bf16*)C_;
                const int bb = mb >> 12;
                const int kv = mb & 4095;
                v8bf vv;
#pragma unroll
                for (int r = 0; r < 8; ++r) vv[r] = (__bf16)acc[ms][ns][r];
                *(v8bf*)(out + (((size_t)(bb * 1024 + n)) << 12) + kv) = vv;
            } else if (OUT_F32) {
                float* out = (float*)C_;
#pragma unroll
                for (int r = 0; r < 8; ++r) out[(size_t)(mb + r) * N + n] = acc[ms][ns][r];
            } else {
                __bf16* out = (__bf16*)C_;
#pragma unroll
                for (int r = 0; r < 8; ++r) out[(size_t)(mb + r) * N + n] = (__bf16)acc[ms][ns][r];
            }
        }
    }
}

// ============================================================================
// Flash attention: grid (B*H=32, Nq/128=16), block 128 (4 waves).
// Each wave owns 32 q-rows (2 16-row subtiles). K/V kv-tiles (32 wide) are
// double-buffered in LDS via async global->LDS b128 copies shared by all 4
// waves. Kp: bf16 [B*4096,1024]; Vt: bf16 [B][1024][4096]; O: bf16 [B*2048,1024]
// ============================================================================
__global__ __launch_bounds__(128)
void attn_wmma_kernel(const float* __restrict__ Q, const __bf16* __restrict__ Kp,
                      const __bf16* __restrict__ Vt, __bf16* __restrict__ O) {
    constexpr int NK = 4096, NQ = 2048, C = 1024;
    __shared__ __bf16 Ks[2][32 * KSTR];   // 9.2 KB
    __shared__ __bf16 Vs[2][64 * VSTR];   // 10.2 KB
    __shared__ __bf16 Pl[4][16 * PSTR];   // 5.1 KB per-wave P staging

    const int tid  = threadIdx.x;
    const int wave = tid >> 5;
    const int lane = tid & 31;
    const int l16  = lane & 15;
    const int g    = lane >> 4;
    const int b    = blockIdx.x >> 4;
    const int h    = blockIdx.x & 15;
    const int q0   = blockIdx.y * 128 + wave * 32;

    const __bf16* kbase = Kp + ((size_t)b * NK) * C + h * 64;
    const __bf16* vbase = Vt + ((size_t)(b * 1024 + h * 64)) * NK;

    // staging coords: Ks tile 32kv x 64d (4KB), Vs tile 64d x 32kv (4KB); 32B/thread each
    const int skv = tid >> 2, sq4 = (tid & 3) * 16;   // K: row skv, d-cols [sq4, sq4+16)
    const int sd  = tid >> 1, sh  = (tid & 1) * 16;   // V: row sd,  kv-cols [sh, sh+16)

    // ---- Q fragments (fp32 -> bf16), 2 subtiles x 2 k-chunks over D=64 ----
    v16bf aq[2][2];
#pragma unroll
    for (int qs = 0; qs < 2; ++qs) {
        const float* qrow = Q + ((size_t)(b * NQ + q0 + qs * 16 + l16)) * C + h * 64;
#pragma unroll
        for (int c = 0; c < 2; ++c) {
            const int dbase = c * 32 + g * 8;
            v4f f0 = *(const v4f*)(qrow + dbase);
            v4f f1 = *(const v4f*)(qrow + dbase + 4);
            v4f f2 = *(const v4f*)(qrow + dbase + 16);
            v4f f3 = *(const v4f*)(qrow + dbase + 20);
            aq[qs][c] = cat16(cvt8(f0, f1), cvt8(f2, f3));
        }
    }

    float mst[2][8], lst[2][8];
#pragma unroll
    for (int qs = 0; qs < 2; ++qs)
#pragma unroll
        for (int r = 0; r < 8; ++r) { mst[qs][r] = -__builtin_inff(); lst[qs][r] = 0.f; }
    v8f acc[2][4];
#pragma unroll
    for (int qs = 0; qs < 2; ++qs)
#pragma unroll
        for (int t = 0; t < 4; ++t) { v8f z = {}; acc[qs][t] = z; }

    __bf16* pl = &Pl[wave][0];

    // issue one kv-tile's async copies (4 x b128 per wave-lane => ASYNCcnt += 4/wave)
    auto issue_tile = [&](int buf, int kv0) {
        async_copy16(LDSOFF(&Ks[buf][skv * KSTR + sq4]),
                     kbase + (size_t)(kv0 + skv) * C + sq4);
        async_copy16(LDSOFF(&Ks[buf][skv * KSTR + sq4 + 8]),
                     kbase + (size_t)(kv0 + skv) * C + sq4 + 8);
        async_copy16(LDSOFF(&Vs[buf][sd * VSTR + sh]),
                     vbase + (size_t)sd * NK + kv0 + sh);
        async_copy16(LDSOFF(&Vs[buf][sd * VSTR + sh + 8]),
                     vbase + (size_t)sd * NK + kv0 + sh + 8);
    };

    issue_tile(0, 0);
    for (int it = 0; it < NK / 32; ++it) {
        const int kv0 = it * 32;
        const int buf = it & 1;
        issue_tile(buf ^ 1, (kv0 + 32) & (NK - 1));  // prefetch next tile (wraps; unused on last)
        wait_async_le4();                            // current tile done; next may be in flight
        __syncthreads();

        // shared K/V fragments for this kv tile
        v16bf bk[2][2];
#pragma unroll
        for (int s = 0; s < 2; ++s)
#pragma unroll
            for (int c = 0; c < 2; ++c) {
                const __bf16* p = &Ks[buf][(s * 16 + l16) * KSTR + c * 32 + g * 8];
                bk[s][c] = cat16(*(const v8bf*)p, *(const v8bf*)(p + 16));
            }
        v16bf bv[4];
#pragma unroll
        for (int t = 0; t < 4; ++t) {
            const __bf16* p = &Vs[buf][(t * 16 + l16) * VSTR + g * 8];
            bv[t] = cat16(*(const v8bf*)p, *(const v8bf*)(p + 16));
        }

#pragma unroll
        for (int qs = 0; qs < 2; ++qs) {
            // S = Q @ K^T
            v8f S[2];
#pragma unroll
            for (int s = 0; s < 2; ++s) {
                v8f cc = {};
                cc = wmma_bf16(aq[qs][0], bk[s][0], cc);
                cc = wmma_bf16(aq[qs][1], bk[s][1], cc);
                S[s] = cc;
            }
            // online softmax (DPP reductions, no LDS traffic)
            float sc[8];
#pragma unroll
            for (int r = 0; r < 8; ++r) {
                float s0 = S[0][r] * 0.125f;  // 1/sqrt(64)
                float s1 = S[1][r] * 0.125f;
                float mx = hmax16(fmaxf(s0, s1));
                float mn = fmaxf(mst[qs][r], mx);
                float e  = __expf(mst[qs][r] - mn);
                float p0 = __expf(s0 - mn);
                float p1 = __expf(s1 - mn);
                float su = hsum16(p0 + p1);
                lst[qs][r] = lst[qs][r] * e + su;
                mst[qs][r] = mn;
                sc[r] = e;
                S[0][r] = p0;
                S[1][r] = p1;
            }
#pragma unroll
            for (int t = 0; t < 4; ++t)
#pragma unroll
                for (int r = 0; r < 8; ++r) acc[qs][t][r] *= sc[r];

            // P: C-layout f32 -> per-wave LDS slab -> A-layout bf16 (16x32)
#pragma unroll
            for (int r = 0; r < 8; ++r) {
                pl[(g * 8 + r) * PSTR + l16]      = (__bf16)S[0][r];
                pl[(g * 8 + r) * PSTR + 16 + l16] = (__bf16)S[1][r];
            }
            const __bf16* pr = pl + l16 * PSTR + g * 8;
            v16bf ap = cat16(*(const v8bf*)pr, *(const v8bf*)(pr + 16));

            // O += P @ V
#pragma unroll
            for (int t = 0; t < 4; ++t) acc[qs][t] = wmma_bf16(ap, bv[t], acc[qs][t]);
        }
        __syncthreads();  // all waves done reading buf before it is refilled
    }

    // ---- normalize and store bf16 [B*NQ, 1024] ----
#pragma unroll
    for (int qs = 0; qs < 2; ++qs) {
        float inv[8];
#pragma unroll
        for (int r = 0; r < 8; ++r) inv[r] = 1.f / lst[qs][r];
#pragma unroll
        for (int t = 0; t < 4; ++t)
#pragma unroll
            for (int r = 0; r < 8; ++r)
                O[(size_t)(b * NQ + q0 + qs * 16 + g * 8 + r) * C + h * 64 + t * 16 + l16] =
                    (__bf16)(acc[qs][t][r] * inv[r]);
    }
}

extern "C" void kernel_launch(void* const* d_in, const int* in_sizes, int n_in,
                              void* d_out, int out_size, void* d_ws, size_t ws_size,
                              hipStream_t stream) {
    (void)in_sizes; (void)n_in; (void)out_size; (void)ws_size;
    const float* q  = (const float*)d_in[0];
    const float* k  = (const float*)d_in[1];
    const float* v  = (const float*)d_in[2];
    const float* Wk = (const float*)d_in[3];
    const float* Wv = (const float*)d_in[4];
    const float* Wo = (const float*)d_in[5];
    float* out = (float*)d_out;

    char* ws = (char*)d_ws;
    __bf16* Kp   = (__bf16*)(ws);                                  // 16 MB [8192,1024]
    __bf16* Vt   = (__bf16*)(ws + (size_t)16 * 1024 * 1024);       // 16 MB [2][1024][4096]
    __bf16* Aout = (__bf16*)(ws + (size_t)32 * 1024 * 1024);       //  8 MB [4096,1024]

    dim3 blk(256);
    gemm_wmma_kernel<0, 0, 0><<<dim3(8, 64), blk, 0, stream>>>(k, Wk, Kp, 8192);
    gemm_wmma_kernel<0, 1, 0><<<dim3(8, 64), blk, 0, stream>>>(v, Wv, Vt, 8192);
    attn_wmma_kernel<<<dim3(32, 16), dim3(128), 0, stream>>>(q, Kp, Vt, Aout);
    gemm_wmma_kernel<1, 0, 1><<<dim3(8, 32), blk, 0, stream>>>(Aout, Wo, out, 4096);
}
